// MRVoxelization_88012469830118
// MI455X (gfx1250) — matrix operand order repaired and checked by reading, not compiled, not run on hardware
//
#include <hip/hip_runtime.h>
#include <math.h>
#include <stdint.h>

// ---- problem constants ----
#define RB   32
#define R3   32768        // RB^3
#define CCH  240          // channels C
#define K2C  480          // 2C (GEMM K)
#define NPT  32768        // points per batch
#define NB   4            // batches
#define NKS  15           // K-steps of 32 (480/32)
#define BN_EPS 1e-5f

typedef __attribute__((ext_vector_type(16))) __bf16 v16bf;
typedef __attribute__((ext_vector_type(8)))  float  v8f;

// ---------------- float atomic max/min via integer punning (deterministic) -----------
__device__ __forceinline__ void atomicMaxF(float* addr, float v) {
  if (v >= 0.0f) atomicMax((int*)addr, __float_as_int(v));
  else           atomicMin((unsigned int*)addr, __float_as_uint(v));
}
__device__ __forceinline__ void atomicMinF(float* addr, float v) {
  if (v >= 0.0f) atomicMin((int*)addr, __float_as_int(v));
  else           atomicMax((unsigned int*)addr, __float_as_uint(v));
}

// ---------------- K0: pre-pack w into per-lane WMMA A-tile bf16 layout ---------------
// wpack[((mt*NKS + ks)*32 + lane)*16 + i] : half i of lane for tile (m-tile mt, k-step ks)
// A 16x32 bf16 layout: M = lane&15, K = ks*32 + (i<8 ? i+8g : i+8+8g), g = lane>>4
__global__ __launch_bounds__(256) void k_pack(const float* __restrict__ w,
                                              __bf16* __restrict__ wpack) {
  const int id = blockIdx.x * 256 + threadIdx.x;   // 16*NKS*32 = 7680
  const int lane = id & 31;
  const int ks = (id >> 5) % NKS;
  const int mt = id / (NKS * 32);
  const int g = lane >> 4;
  const int M = mt * 16 + (lane & 15);
  v16bf out;
#pragma unroll
  for (int i = 0; i < 16; ++i) {
    const int kk = ks * 32 + ((i < 8) ? (i + 8 * g) : (i + 8 + 8 * g));
    const float f = (M < CCH) ? w[(size_t)M * K2C + kk] : 0.0f;
    out[i] = (__bf16)f;
  }
  *(v16bf*)(wpack + ((size_t)id << 4)) = out;
}

// ---------------- K1: per (b,dim) mean of coords -------------------------------------
__global__ __launch_bounds__(256) void k_mean(const float* __restrict__ coords,
                                              float* __restrict__ meanBuf) {
  const int bd = blockIdx.x;                 // 0..11  (b*3+d)
  const float* p = coords + (size_t)bd * NPT;
  float s = 0.0f;
  for (int n = threadIdx.x; n < NPT; n += 256) s += p[n];
  __shared__ float red[256];
  red[threadIdx.x] = s; __syncthreads();
  for (int off = 128; off > 0; off >>= 1) {
    if (threadIdx.x < off) red[threadIdx.x] += red[threadIdx.x + off];
    __syncthreads();
  }
  if (threadIdx.x == 0) meanBuf[bd] = red[0] * (1.0f / (float)NPT);
}

// ---------------- K2: per-batch scale = 2*max||coord-mean|| --------------------------
__global__ __launch_bounds__(256) void k_scale(const float* __restrict__ coords,
                                               const float* __restrict__ meanBuf,
                                               float* __restrict__ scaleBuf) {
  const int b = blockIdx.x;
  const float mx = meanBuf[b * 3 + 0], my = meanBuf[b * 3 + 1], mz = meanBuf[b * 3 + 2];
  const float* px = coords + ((size_t)b * 3 + 0) * NPT;
  const float* py = coords + ((size_t)b * 3 + 1) * NPT;
  const float* pz = coords + ((size_t)b * 3 + 2) * NPT;
  float m = 0.0f;
  for (int n = threadIdx.x; n < NPT; n += 256) {
    float dx = px[n] - mx, dy = py[n] - my, dz = pz[n] - mz;
    m = fmaxf(m, dx * dx + dy * dy + dz * dz);
  }
  __shared__ float red[256];
  red[threadIdx.x] = m; __syncthreads();
  for (int off = 128; off > 0; off >>= 1) {
    if (threadIdx.x < off) red[threadIdx.x] = fmaxf(red[threadIdx.x], red[threadIdx.x + off]);
    __syncthreads();
  }
  if (threadIdx.x == 0) scaleBuf[b] = 2.0f * sqrtf(red[0]);
}

// ---------------- K3: normalized coords (output) + flat voxel id ---------------------
__global__ __launch_bounds__(256) void k_vox(const float* __restrict__ coords,
                                             const float* __restrict__ meanBuf,
                                             const float* __restrict__ scaleBuf,
                                             float* __restrict__ ncOut,
                                             int* __restrict__ pos) {
  const int idx = blockIdx.x * 256 + threadIdx.x;   // b*NPT + n
  const int b = idx >> 15, n = idx & (NPT - 1);
  const float inv = 1.0f / scaleBuf[b];
  int vx[3];
#pragma unroll
  for (int d = 0; d < 3; ++d) {
    float v = (coords[((size_t)b * 3 + d) * NPT + n] - meanBuf[b * 3 + d]) * inv + 0.5f;
    v = v * (float)RB;
    v = fminf(fmaxf(v, 0.0f), (float)(RB - 1));
    ncOut[((size_t)b * 3 + d) * NPT + n] = v;       // reference returns clipped nc
    vx[d] = (int)rintf(v);                          // round-half-even == jnp.round
  }
  pos[idx] = vx[0] + vx[1] * RB + vx[2] * RB * RB;
}

// ---------------- K4: init fea: max half -> -inf, min half -> +inf -------------------
__global__ __launch_bounds__(256) void k_init(unsigned int* __restrict__ fea) {
  const size_t idx = (size_t)blockIdx.x * 256 + threadIdx.x;   // NB*K2C*R3
  const int k = (int)((idx >> 15) % K2C);
  fea[idx] = (k < CCH) ? 0xFF800000u : 0x7F800000u;
}

// ---------------- K5: voxel scatter (segment max / min) ------------------------------
__global__ __launch_bounds__(256) void k_scatter(const float* __restrict__ features,
                                                 const int* __restrict__ pos,
                                                 float* __restrict__ fea) {
  const int idx = blockIdx.x * 256 + threadIdx.x;   // b*NPT + n
  const int b = idx >> 15, n = idx & (NPT - 1);
  const int p = pos[idx];
  float* fmax = fea + (size_t)b * K2C * R3 + p;
  float* fmin = fmax + (size_t)CCH * R3;
  const float* f = features + (size_t)b * CCH * NPT + n;
  for (int c = 0; c < CCH; ++c) {
    float v = f[(size_t)c * NPT];
    atomicMaxF(fmax + (size_t)c * R3, v);
    atomicMinF(fmin + (size_t)c * R3, v);
  }
}

// ---------------- K6: bf16 WMMA GEMM  y[b,c,v] = w[c,:]·fea[b,:,v] + bias[c] ---------
// Workgroup (8 waves) covers full M=240 (2 m-tiles/wave) x 64 voxels.
// Next K=32 x V=64 f32 panel fetched with gfx1250 async-to-LDS DMA while current
// panel's WMMAs run; then masked (±inf -> 0), cvt to bf16, stored pre-swizzled into
// the WMMA B register layout (per-lane contiguous 32B -> two ds_load_b128 per tile).
__device__ __forceinline__ void stage_issue(const float* __restrict__ src,
                                            unsigned rawOff, int tid) {
#pragma unroll
  for (int j = 0; j < 2; ++j) {
    const int q  = tid + j * 256;            // 512 x b128 chunks
    const int kk = q >> 4;                   // panel row 0..31
    const int v4 = (q & 15) * 4;             // col 0..60 step 4
    const unsigned dst = rawOff + (unsigned)((kk * 64 + v4) * 4);
    const unsigned long long ga =
        (unsigned long long)(uintptr_t)(src + (size_t)kk * R3 + v4);
    asm volatile("global_load_async_to_lds_b128 %0, %1, off"
                 :: "v"(dst), "v"(ga) : "memory");
  }
}

__device__ __forceinline__ void stage_commit(const float* __restrict__ raw,
                                             __bf16* __restrict__ dstPanel, int tid) {
  asm volatile("s_wait_asynccnt 0x0" ::: "memory");
#pragma unroll
  for (int j = 0; j < 2; ++j) {
    const int q  = tid + j * 256;
    const int kk = q >> 4;
    const int v4 = (q & 15) * 4;
    const float4 f4 = *(const float4*)(raw + kk * 64 + v4);
    const int i  = (kk & 7) | ((kk >> 1) & 8); // half index within lane chunk
    const int lh = (kk & 8) << 1;            // lane half-group bit
    const int vt = v4 >> 4;                  // 16-col tile
    const float fv[4] = {f4.x, f4.y, f4.z, f4.w};
#pragma unroll
    for (int u = 0; u < 4; ++u) {
      float f = fv[u];
      if (fabsf(f) > 3.0e38f) f = 0.0f;      // empty voxel (±inf sentinel) -> 0
      const int L = ((v4 + u) & 15) | lh;    // consuming lane
      dstPanel[vt * 512 + L * 16 + i] = (__bf16)f;
    }
  }
}

__global__ __launch_bounds__(256) void k_gemm(const __bf16* __restrict__ wpack,
                                              const float* __restrict__ bias,
                                              const float* __restrict__ fea,
                                              float* __restrict__ y) {
  const int b  = blockIdx.x >> 9;          // 512 v-groups per batch
  const int v0 = (blockIdx.x & 511) * 64;
  const int tid  = threadIdx.x;
  const int wave = tid >> 5;
  const int lane = tid & 31;
  const int g  = lane >> 4;
  const int ln = lane & 15;

  __shared__ __align__(16) float  raw[32 * 64];       // async landing zone (8 KB)
  __shared__ __align__(32) __bf16 panel[2][4 * 512];  // swizzled bf16, double buffered

  const unsigned rawOff = (unsigned)(uintptr_t)&raw[0];

  v8f acc[2][4];
#pragma unroll
  for (int t = 0; t < 2; ++t)
#pragma unroll
    for (int vt = 0; vt < 4; ++vt) acc[t][vt] = (v8f)0.0f;

  const float* feaB = fea + (size_t)b * K2C * R3 + v0;

  // prologue: stage panel 0
  stage_issue(feaB, rawOff, tid);
  stage_commit(raw, &panel[0][0], tid);
  __syncthreads();

  for (int ks = 0; ks < NKS; ++ks) {
    const int cur = ks & 1;
    if (ks + 1 < NKS)                        // kick off async DMA of next panel
      stage_issue(feaB + (size_t)(ks + 1) * 32 * R3, rawOff, tid);

    // B tiles: per-lane contiguous 32B chunk -> two ds_load_b128 each
    v16bf bm[4];
#pragma unroll
    for (int vt = 0; vt < 4; ++vt)
      bm[vt] = *(const v16bf*)(&panel[cur][vt * 512 + lane * 16]);

#pragma unroll
    for (int t = 0; t < 2; ++t) {
      const int mt = wave * 2 + t;
      const v16bf a = *(const v16bf*)(wpack +
                        ((size_t)((mt * NKS + ks) * 32 + lane) << 4));
#pragma unroll
      for (int vt = 0; vt < 4; ++vt)
        acc[t][vt] = __builtin_amdgcn_wmma_f32_16x16x32_bf16(
            false, a, false, bm[vt], (short)0, acc[t][vt], false, false);
    }

    if (ks + 1 < NKS)                        // convert+swizzle next panel
      stage_commit(raw, &panel[cur ^ 1][0], tid);
    __syncthreads();
  }

  // store pre-BN y (+bias). D layout: lane ln = N col, vgpr r -> M = 16*mt + r + 8*g
  float* yB = y + (size_t)b * CCH * R3;
#pragma unroll
  for (int t = 0; t < 2; ++t) {
    const int mt = wave * 2 + t;
    if (mt * 16 >= CCH) continue;            // padded 16th m-tile
#pragma unroll
    for (int vt = 0; vt < 4; ++vt)
#pragma unroll
      for (int r = 0; r < 8; ++r) {
        const int m = mt * 16 + r + 8 * g;
        yB[(size_t)m * R3 + v0 + vt * 16 + ln] = acc[t][vt][r] + bias[m];
      }
  }
}

// ---------------- K7: per-channel batch stats (deterministic tree reduction) ---------
__global__ __launch_bounds__(256) void k_stats(const float* __restrict__ y,
                                               float* __restrict__ chanMean,
                                               float* __restrict__ chanRstd) {
  const int c = blockIdx.x;                // 0..239
  float s = 0.0f, ss = 0.0f;
  for (int idx = threadIdx.x; idx < NB * R3; idx += 256) {
    const int b = idx >> 15, v = idx & (R3 - 1);
    const float val = y[((size_t)b * CCH + c) * R3 + v];
    s += val; ss += val * val;
  }
  __shared__ float rs[256], rq[256];
  rs[threadIdx.x] = s; rq[threadIdx.x] = ss; __syncthreads();
  for (int off = 128; off > 0; off >>= 1) {
    if (threadIdx.x < off) {
      rs[threadIdx.x] += rs[threadIdx.x + off];
      rq[threadIdx.x] += rq[threadIdx.x + off];
    }
    __syncthreads();
  }
  if (threadIdx.x == 0) {
    const float invM = 1.0f / (float)(NB * R3);
    const float mean = rs[0] * invM;
    const float var  = rq[0] * invM - mean * mean;   // biased, matches jnp.var
    chanMean[c] = mean;
    chanRstd[c] = rsqrtf(var + BN_EPS);
  }
}

// ---------------- K8: BN(affine) + Swish, in place on d_out --------------------------
__global__ __launch_bounds__(256) void k_bn_swish(float* __restrict__ y,
                                                  const float* __restrict__ chanMean,
                                                  const float* __restrict__ chanRstd,
                                                  const float* __restrict__ gamma,
                                                  const float* __restrict__ beta) {
  const size_t idx = (size_t)blockIdx.x * 256 + threadIdx.x;  // NB*CCH*R3
  const int c = (int)((idx >> 15) % CCH);
  float v = y[idx];
  v = gamma[c] * (v - chanMean[c]) * chanRstd[c] + beta[c];
  y[idx] = v * (1.0f / (1.0f + expf(-v)));
}

// -------------------------------------------------------------------------------------
extern "C" void kernel_launch(void* const* d_in, const int* in_sizes, int n_in,
                              void* d_out, int out_size, void* d_ws, size_t ws_size,
                              hipStream_t stream) {
  const float* features = (const float*)d_in[0];   // [4,240,32768]
  const float* coords   = (const float*)d_in[1];   // [4,3,32768]
  const float* w        = (const float*)d_in[2];   // [240,480]
  const float* bias     = (const float*)d_in[3];   // [240]
  const float* gamma    = (const float*)d_in[4];   // [240]
  const float* beta     = (const float*)d_in[5];   // [240]

  float* y     = (float*)d_out;                            // [4,240,32768]
  float* ncOut = (float*)d_out + (size_t)NB * CCH * R3;    // [4,3,32768]

  // workspace layout (floats)
  float*  wsf      = (float*)d_ws;
  float*  meanBuf  = wsf;                          // 12
  float*  scaleBuf = wsf + 12;                     // 4
  float*  chanMean = wsf + 16;                     // 240
  float*  chanRstd = wsf + 256;                    // 240
  __bf16* wpack    = (__bf16*)(wsf + 512);         // 16*15*32*16 bf16 = 61440 floats
  int*    pos      = (int*)(wsf + 512 + 61440);    // 131072 ints
  float*  fea      = wsf + 512 + 61440 + (size_t)NB * NPT;  // [4,480,32768] = 252 MB

  k_pack   <<<(16 * NKS * 32) / 256, 256, 0, stream>>>(w, wpack);
  k_mean   <<<NB * 3, 256, 0, stream>>>(coords, meanBuf);
  k_scale  <<<NB,     256, 0, stream>>>(coords, meanBuf, scaleBuf);
  k_vox    <<<(NB * NPT) / 256, 256, 0, stream>>>(coords, meanBuf, scaleBuf, ncOut, pos);
  k_init   <<<(int)(((size_t)NB * K2C * R3) / 256), 256, 0, stream>>>((unsigned int*)fea);
  k_scatter<<<(NB * NPT) / 256, 256, 0, stream>>>(features, pos, fea);
  k_gemm   <<<NB * 512, 256, 0, stream>>>(wpack, bias, fea, y);
  k_stats  <<<CCH, 256, 0, stream>>>(y, chanMean, chanRstd);
  k_bn_swish<<<(int)(((size_t)NB * CCH * R3) / 256), 256, 0, stream>>>(y, chanMean, chanRstd, gamma, beta);
}
